// SE3EquivariantConv_40836549050449
// MI455X (gfx1250) — compile-verified, matrix operand order checked by reading, not run on hardware
//
#include <hip/hip_runtime.h>
#include <hip/hip_bf16.h>
#include <math.h>

typedef __bf16 bf16;
typedef bf16  bf16x4  __attribute__((ext_vector_type(4)));
typedef bf16  bf16x8  __attribute__((ext_vector_type(8)));
typedef bf16  bf16x16 __attribute__((ext_vector_type(16)));
typedef float v8f     __attribute__((ext_vector_type(8)));

#define HDIM 256
#define MT   64      // rows (edges/nodes) per block
#define NTHREADS 256 // 8 waves: 2 M-groups x 4 N-groups
#define LDA  520     // bf16 row stride for K=512 staging (pad vs 512 to spread LDS banks)
#define LDX  264     // bf16 row stride for K=256 tiles

__device__ __forceinline__ v8f vzero8() {
  v8f z = {0.f,0.f,0.f,0.f,0.f,0.f,0.f,0.f};
  return z;
}

__device__ __forceinline__ float silu_f(float x) {
  return x * (1.0f / (1.0f + __expf(-x)));
}

// ---------------------------------------------------------------------------
// Pack a (K x 256) f32 weight into bf16 WMMA B-fragments.
// Fragment (kt,nt) covers K rows [kt*32, kt*32+32), N cols [nt*16, nt*16+16).
// ISA 16-bit operand layout: lanes 0-15 hold K {0..7, 16..23}, lanes 16-31
// hold K {8..15, 24..31}; element e -> k = e + (lane>=16)*8 + (e>=8)*8.
// dst[((kt*16+nt)*32 + lane)*16 + e]  => each lane reads 32 contiguous bytes.
// ---------------------------------------------------------------------------
__global__ void pack_weights(const float* __restrict__ src,
                             bf16* __restrict__ dst, int KT) {
  int idx = blockIdx.x * 256 + threadIdx.x;
  int total = KT * 16 * 32 * 16;
  if (idx >= total) return;
  int e    = idx & 15;
  int lane = (idx >> 4) & 31;
  int frag = idx >> 9;
  int nt   = frag & 15;
  int kt   = frag >> 4;
  int kh   = (lane >> 4) << 3;
  int k    = kt * 32 + e + kh + ((e >= 8) ? 8 : 0);
  int n    = nt * 16 + (lane & 15);
  dst[idx] = (bf16)src[(size_t)k * HDIM + n];
}

// ---------------------------------------------------------------------------
// Per-wave 32xKx64 GEMM slice of the block's 64xKx256 tile.
// A (bf16, LDS, row stride lda, K mult of 32); B packed fragments in global.
// Wave (mg,ng) covers rows [mg*32, mg*32+32), cols [ng*64, ng*64+64).
// acc[ms][nt]: 16x16 f32 C tile, rows mg*32+ms*16.., cols (ng*4+nt)*16...
// ---------------------------------------------------------------------------
__device__ __forceinline__ void gemm32xKx64(
    const bf16* __restrict__ lA, int lda, int K,
    const bf16* __restrict__ Bp, int lane, int mg, int ng, v8f acc[2][4]) {
  const int m  = lane & 15;
  const int kh = (lane >> 4) << 3;
  const int KT = K >> 5;
  for (int kt = 0; kt < KT; ++kt) {
    bf16x16 afr[2];
#pragma unroll
    for (int ms = 0; ms < 2; ++ms) {
      const bf16* ap = lA + (mg * 32 + ms * 16 + m) * lda + (kt << 5) + kh;
      bf16x8 lo = *(const bf16x8*)ap;            // K = kt*32 + kh + 0..7
      bf16x8 hi = *(const bf16x8*)(ap + 16);     // K = kt*32 + 16 + kh + 0..7
      afr[ms] = __builtin_shufflevector(lo, hi, 0,1,2,3,4,5,6,7,
                                                 8,9,10,11,12,13,14,15);
    }
#pragma unroll
    for (int nt = 0; nt < 4; ++nt) {
      const bf16x16 bfr = *(const bf16x16*)(
          Bp + (((size_t)(kt * 16 + ng * 4 + nt)) * 32 + lane) * 16);
#pragma unroll
      for (int ms = 0; ms < 2; ++ms)
        acc[ms][nt] = __builtin_amdgcn_wmma_f32_16x16x32_bf16(
            false, afr[ms], false, bfr, (short)0, acc[ms][nt], false, false);
    }
  }
}

// ---------------------------------------------------------------------------
// Edge kernel: per block, 64 edges through edge-MLP and pos-MLP.
//   hidden    = silu([hL|hR] @ ew1[0:512] + dist * ew1[512] + eb1)
//   edge_attr = hidden @ ew2 + eb2                     -> ea_out (bf16)
//   t         = silu(edge_attr @ pw1 + pb1)
//   delta_pos = t @ pw2                                -> dp_out (f32)
// ---------------------------------------------------------------------------
__global__ __launch_bounds__(NTHREADS) void se3_edge_kernel(
    const float* __restrict__ h, const float* __restrict__ pos,
    const float* __restrict__ ew1f, const float* __restrict__ eb1,
    const float* __restrict__ eb2,  const float* __restrict__ pb1,
    const float* __restrict__ pw2,
    const bf16* __restrict__ ew1p, const bf16* __restrict__ ew2p,
    const bf16* __restrict__ pw1p,
    bf16* __restrict__ ea_out, float* __restrict__ dp_out, int n_edges) {
  __shared__ float lds_big[MT * LDX];   // 66 KB: bf16 A (stride LDA) then f32 t
  __shared__ bf16  lds_x[MT * LDX];     // hidden (bf16)
  __shared__ bf16  lds_y[MT * LDX];     // edge_attr (bf16)
  __shared__ float lds_dist[MT];

  bf16*  lds_a = (bf16*)lds_big;
  float* lds_t = lds_big;

  const int tid   = threadIdx.x;
  const int lane  = tid & 31;
  const int wave  = tid >> 5;
  const int mg    = wave >> 2;
  const int ng    = wave & 3;
  const int eBase = blockIdx.x * MT;

  if (tid < MT) {
    int e = eBase + tid;
    float d = 0.f;
    if (e < n_edges) {
      float dx = pos[3 * e + 3] - pos[3 * e + 0];
      float dy = pos[3 * e + 4] - pos[3 * e + 1];
      float dz = pos[3 * e + 5] - pos[3 * e + 2];
      d = sqrtf(dx * dx + dy * dy + dz * dz);
    }
    lds_dist[tid] = d;
  }

  // Stage A rows: [h[e] | h[e+1]] as bf16 (512 wide, stride LDA)
  for (int it = 0; it < 32; ++it) {
    int idx = (it * NTHREADS + tid) * 4;      // 64*512 = 32768 elems
    int r = idx >> 9, c = idx & 511;
    int e = eBase + r;
    float4 v = make_float4(0.f, 0.f, 0.f, 0.f);
    if (e < n_edges) {
      int node = (c < HDIM) ? e : (e + 1);
      v = *(const float4*)(h + (size_t)node * HDIM + (c & (HDIM - 1)));
    }
    bf16x4 bv = {(bf16)v.x, (bf16)v.y, (bf16)v.z, (bf16)v.w};
    *(bf16x4*)(lds_a + r * LDA + c) = bv;
  }
  __syncthreads();

  const int nl   = lane & 15;
  const int mofs = (lane >> 4) << 3;
  v8f acc[2][4];

  // -------- stage 1: hidden --------
  for (int ms = 0; ms < 2; ++ms)
    for (int nt = 0; nt < 4; ++nt) acc[ms][nt] = vzero8();
  gemm32xKx64(lds_a, LDA, 512, ew1p, lane, mg, ng, acc);
  const float* ew1d = ew1f + (size_t)512 * HDIM;   // rank-1 dist column
#pragma unroll
  for (int ms = 0; ms < 2; ++ms)
#pragma unroll
    for (int nt = 0; nt < 4; ++nt) {
      int n = ng * 64 + nt * 16 + nl;
      float wd = ew1d[n], b = eb1[n];
#pragma unroll
      for (int j = 0; j < 8; ++j) {
        int mr = mg * 32 + ms * 16 + mofs + j;
        float x = acc[ms][nt][j] + lds_dist[mr] * wd + b;
        lds_x[mr * LDX + n] = (bf16)silu_f(x);
      }
    }
  __syncthreads();

  // -------- stage 2: edge_attr --------
  for (int ms = 0; ms < 2; ++ms)
    for (int nt = 0; nt < 4; ++nt) acc[ms][nt] = vzero8();
  gemm32xKx64(lds_x, LDX, 256, ew2p, lane, mg, ng, acc);
#pragma unroll
  for (int ms = 0; ms < 2; ++ms)
#pragma unroll
    for (int nt = 0; nt < 4; ++nt) {
      int n = ng * 64 + nt * 16 + nl;
      float b = eb2[n];
#pragma unroll
      for (int j = 0; j < 8; ++j) {
        int mr = mg * 32 + ms * 16 + mofs + j;
        lds_y[mr * LDX + n] = (bf16)(acc[ms][nt][j] + b);
      }
    }
  __syncthreads();

  // coalesced copy edge_attr -> global (bf16), skipping pad columns
  for (int it = 0; it < 8; ++it) {
    int chunk = it * NTHREADS + tid;     // 2048 chunks of 8 bf16 (16 B)
    int r = chunk >> 5;
    int c = (chunk & 31) * 8;
    if (eBase + r < n_edges)
      *(uint4*)(ea_out + (size_t)(eBase + r) * HDIM + c) =
          *(const uint4*)(lds_y + r * LDX + c);
  }

  // -------- stage 3: t = silu(edge_attr @ pw1 + pb1) --------
  for (int ms = 0; ms < 2; ++ms)
    for (int nt = 0; nt < 4; ++nt) acc[ms][nt] = vzero8();
  gemm32xKx64(lds_y, LDX, 256, pw1p, lane, mg, ng, acc);
#pragma unroll
  for (int ms = 0; ms < 2; ++ms)
#pragma unroll
    for (int nt = 0; nt < 4; ++nt) {
      int n = ng * 64 + nt * 16 + nl;
      float b = pb1[n];
#pragma unroll
      for (int j = 0; j < 8; ++j) {
        int mr = mg * 32 + ms * 16 + mofs + j;
        lds_t[mr * LDX + n] = silu_f(acc[ms][nt][j] + b);
      }
    }
  __syncthreads();

  // -------- stage 4: delta_pos = t @ pw2 (256x3, VALU) --------
  if (tid < 3 * MT) {
    int r = tid / 3, j = tid - 3 * (tid / 3);
    float s = 0.f;
    for (int k = 0; k < HDIM; ++k) s += lds_t[r * LDX + k] * pw2[k * 3 + j];
    int e = eBase + r;
    if (e < n_edges) dp_out[(size_t)e * 3 + j] = s;
  }
}

// ---------------------------------------------------------------------------
// Node kernel: per block, 64 nodes.
//   node_update[i] = ea[i] + ea[i-1]  (chain scatter-add == shifted reads)
//   h_out = silu([h|node_update] @ nw1 + nb1) @ nw2 + nb2
//   pos_out = pos + 0.1 * (dp[i] - dp[i-1])
// ---------------------------------------------------------------------------
__global__ __launch_bounds__(NTHREADS) void se3_node_kernel(
    const float* __restrict__ h, const float* __restrict__ pos,
    const float* __restrict__ nb1, const float* __restrict__ nb2,
    const bf16* __restrict__ nw1p, const bf16* __restrict__ nw2p,
    const bf16* __restrict__ ea, const float* __restrict__ dp,
    float* __restrict__ h_out, float* __restrict__ pos_out, int n_nodes) {
  __shared__ bf16 lds_a[MT * LDA];      // 65 KB
  __shared__ bf16 lds_x[MT * LDX];      // 33 KB

  const int tid     = threadIdx.x;
  const int lane    = tid & 31;
  const int wave    = tid >> 5;
  const int mg      = wave >> 2;
  const int ng      = wave & 3;
  const int iBase   = blockIdx.x * MT;
  const int n_edges = n_nodes - 1;

  // pos update (independent of GEMM pipeline)
  if (tid < 3 * MT) {
    int r = tid / 3, j = tid - 3 * (tid / 3);
    int i = iBase + r;
    if (i < n_nodes) {
      float u = 0.f;
      if (i < n_edges) u += dp[(size_t)i * 3 + j];
      if (i > 0)       u -= dp[(size_t)(i - 1) * 3 + j];
      pos_out[(size_t)i * 3 + j] = pos[(size_t)i * 3 + j] + 0.1f * u;
    }
  }

  // Stage A rows: [h[i] | ea[i]+ea[i-1]] as bf16
  for (int it = 0; it < 32; ++it) {
    int idx = (it * NTHREADS + tid) * 4;
    int r = idx >> 9, c = idx & 511;
    int i = iBase + r;
    bf16x4 bv = {(bf16)0.f, (bf16)0.f, (bf16)0.f, (bf16)0.f};
    if (i < n_nodes) {
      if (c < HDIM) {
        float4 v = *(const float4*)(h + (size_t)i * HDIM + c);
        bv = (bf16x4){(bf16)v.x, (bf16)v.y, (bf16)v.z, (bf16)v.w};
      } else {
        int cc = c & (HDIM - 1);
        float s0 = 0.f, s1 = 0.f, s2 = 0.f, s3 = 0.f;
        if (i < n_edges) {
          bf16x4 a = *(const bf16x4*)(ea + (size_t)i * HDIM + cc);
          s0 += (float)a[0]; s1 += (float)a[1];
          s2 += (float)a[2]; s3 += (float)a[3];
        }
        if (i > 0) {
          bf16x4 b = *(const bf16x4*)(ea + (size_t)(i - 1) * HDIM + cc);
          s0 += (float)b[0]; s1 += (float)b[1];
          s2 += (float)b[2]; s3 += (float)b[3];
        }
        bv = (bf16x4){(bf16)s0, (bf16)s1, (bf16)s2, (bf16)s3};
      }
    }
    *(bf16x4*)(lds_a + r * LDA + c) = bv;
  }
  __syncthreads();

  const int nl   = lane & 15;
  const int mofs = (lane >> 4) << 3;
  v8f acc[2][4];

  // -------- stage 1: silu(A @ nw1 + nb1) --------
  for (int ms = 0; ms < 2; ++ms)
    for (int nt = 0; nt < 4; ++nt) acc[ms][nt] = vzero8();
  gemm32xKx64(lds_a, LDA, 512, nw1p, lane, mg, ng, acc);
#pragma unroll
  for (int ms = 0; ms < 2; ++ms)
#pragma unroll
    for (int nt = 0; nt < 4; ++nt) {
      int n = ng * 64 + nt * 16 + nl;
      float b = nb1[n];
#pragma unroll
      for (int j = 0; j < 8; ++j) {
        int mr = mg * 32 + ms * 16 + mofs + j;
        lds_x[mr * LDX + n] = (bf16)silu_f(acc[ms][nt][j] + b);
      }
    }
  __syncthreads();

  // -------- stage 2: @ nw2 + nb2 -> h_out --------
  for (int ms = 0; ms < 2; ++ms)
    for (int nt = 0; nt < 4; ++nt) acc[ms][nt] = vzero8();
  gemm32xKx64(lds_x, LDX, 256, nw2p, lane, mg, ng, acc);
#pragma unroll
  for (int ms = 0; ms < 2; ++ms)
#pragma unroll
    for (int nt = 0; nt < 4; ++nt) {
      int n = ng * 64 + nt * 16 + nl;
      float b = nb2[n];
#pragma unroll
      for (int j = 0; j < 8; ++j) {
        int mr = mg * 32 + ms * 16 + mofs + j;
        int i = iBase + mr;
        if (i < n_nodes)
          h_out[(size_t)i * HDIM + n] = acc[ms][nt][j] + b;
      }
    }
}

// ---------------------------------------------------------------------------
extern "C" void kernel_launch(void* const* d_in, const int* in_sizes, int n_in,
                              void* d_out, int out_size, void* d_ws,
                              size_t ws_size, hipStream_t stream) {
  const float* h   = (const float*)d_in[0];
  const float* pos = (const float*)d_in[1];
  const float* ew1 = (const float*)d_in[2];
  const float* eb1 = (const float*)d_in[3];
  const float* ew2 = (const float*)d_in[4];
  const float* eb2 = (const float*)d_in[5];
  const float* pw1 = (const float*)d_in[6];
  const float* pb1 = (const float*)d_in[7];
  const float* pw2 = (const float*)d_in[8];
  const float* nw1 = (const float*)d_in[9];
  const float* nb1 = (const float*)d_in[10];
  const float* nw2 = (const float*)d_in[11];
  const float* nb2 = (const float*)d_in[12];

  const int N       = in_sizes[0] / HDIM;
  const int n_edges = N - 1;

  // workspace layout
  char* ws = (char*)d_ws;
  size_t o = 0;
  bf16* ew1p = (bf16*)(ws + o); o += (size_t)16 * 16 * 512 * 2;  // 512x256
  bf16* ew2p = (bf16*)(ws + o); o += (size_t)8  * 16 * 512 * 2;  // 256x256
  bf16* pw1p = (bf16*)(ws + o); o += (size_t)8  * 16 * 512 * 2;  // 256x256
  bf16* nw1p = (bf16*)(ws + o); o += (size_t)16 * 16 * 512 * 2;  // 512x256
  bf16* nw2p = (bf16*)(ws + o); o += (size_t)8  * 16 * 512 * 2;  // 256x256
  bf16* ea   = (bf16*)(ws + o); o += (size_t)N * HDIM * 2;       // edge_attr
  float* dp  = (float*)(ws + o); o += (size_t)N * 3 * 4;         // delta_pos

  float* h_out   = (float*)d_out;
  float* pos_out = h_out + (size_t)N * HDIM;

  pack_weights<<<512, 256, 0, stream>>>(ew1, ew1p, 16);  // uses rows 0..511
  pack_weights<<<256, 256, 0, stream>>>(ew2, ew2p, 8);
  pack_weights<<<256, 256, 0, stream>>>(pw1, pw1p, 8);
  pack_weights<<<512, 256, 0, stream>>>(nw1, nw1p, 16);
  pack_weights<<<256, 256, 0, stream>>>(nw2, nw2p, 8);

  int gA = (n_edges + MT - 1) / MT;
  se3_edge_kernel<<<gA, NTHREADS, 0, stream>>>(h, pos, ew1, eb1, eb2, pb1, pw2,
                                               ew1p, ew2p, pw1p, ea, dp,
                                               n_edges);
  int gB = (N + MT - 1) / MT;
  se3_node_kernel<<<gB, NTHREADS, 0, stream>>>(h, pos, nb1, nb2, nw1p, nw2p,
                                               ea, dp, h_out, pos_out, N);
}
